// EquivariantTransformerLayer_83313775608018
// MI455X (gfx1250) — matrix operand order repaired and checked by reading, not compiled, not run on hardware
//
#include <hip/hip_runtime.h>

typedef __attribute__((ext_vector_type(16))) _Float16 v16h;
typedef __attribute__((ext_vector_type(8)))  _Float16 v8h;
typedef __attribute__((ext_vector_type(8)))  float    v8f;

#define N_TOK 1024
#define DM 64

__device__ __forceinline__ float fast_sigmoid(float x) {
  // v_exp_f32 + v_add + v_rcp_f32 (avoid IEEE div expansion)
  return __builtin_amdgcn_rcpf(1.f + __expf(-x));
}
__device__ __forceinline__ float fast_silu(float x) {
  return x * fast_sigmoid(x);
}

// ---------------------------------------------------------------------------
// K0: precompose Wc = edge_w2 @ coord_w1 (fp16, transposed [n][k]),
//     bc = edge_b2 @ coord_w1 + coord_b1, and w1T fp16 [64][16] (K padded 9->16)
// ---------------------------------------------------------------------------
__global__ void k_prepw(const float* __restrict__ edge_w2, const float* __restrict__ coord_w1,
                        const float* __restrict__ edge_b2, const float* __restrict__ coord_b1,
                        const float* __restrict__ edge_w1,
                        _Float16* __restrict__ WcT, float* __restrict__ bc,
                        _Float16* __restrict__ w1T) {
  int tid = threadIdx.x;
  for (int idx = tid; idx < 64 * 64; idx += 256) {
    int n = idx >> 6, m = idx & 63;
    float acc = 0.f;
    for (int k = 0; k < 64; ++k) acc += edge_w2[m * 64 + k] * coord_w1[k * 64 + n];
    WcT[n * 64 + m] = (_Float16)acc;   // [n][K=m]
  }
  if (tid < 64) {
    float acc = coord_b1[tid];
    for (int k = 0; k < 64; ++k) acc += edge_b2[k] * coord_w1[k * 64 + tid];
    bc[tid] = acc;
  }
  for (int idx = tid; idx < 64 * 16; idx += 256) {
    int n = idx >> 4, k = idx & 15;
    w1T[idx] = (_Float16)((k < 9) ? edge_w1[k * 64 + n] : 0.f);
  }
}

// ---------------------------------------------------------------------------
// K1: LayerNorm1 (one wave per row, 2 elems per lane)
// ---------------------------------------------------------------------------
__global__ void k_ln1(const float* __restrict__ feats, const float* __restrict__ g,
                      const float* __restrict__ b, float* __restrict__ x) {
  int lane = threadIdx.x & 31, wv = threadIdx.x >> 5;
  int row = blockIdx.x * 8 + wv;
  float x0 = feats[row * 64 + lane];
  float x1 = feats[row * 64 + 32 + lane];
  float s = x0 + x1;
#pragma unroll
  for (int m = 1; m < 32; m <<= 1) s += __shfl_xor(s, m);
  float mean = s * (1.f / 64.f);
  float d0 = x0 - mean, d1 = x1 - mean;
  float v = d0 * d0 + d1 * d1;
#pragma unroll
  for (int m = 1; m < 32; m <<= 1) v += __shfl_xor(v, m);
  float inv = rsqrtf(v * (1.f / 64.f) + 1e-5f);
  x[row * 64 + lane]      = d0 * inv * g[lane]      + b[lane];
  x[row * 64 + 32 + lane] = d1 * inv * g[lane + 32] + b[lane + 32];
}

// ---------------------------------------------------------------------------
// K2: qkv = x @ qkv_w    (block per row, 192 outputs)
// ---------------------------------------------------------------------------
__global__ void k_qkv(const float* __restrict__ x, const float* __restrict__ w,
                      float* __restrict__ qkv) {
  __shared__ float xs[64];
  int i = blockIdx.x, t = threadIdx.x;
  if (t < 64) xs[t] = x[i * 64 + t];
  __syncthreads();
  if (t < 192) {
    float acc = 0.f;
#pragma unroll
    for (int m = 0; m < 64; ++m) acc += xs[m] * w[m * 192 + t];
    qkv[i * 192 + t] = acc;
  }
}

// ---------------------------------------------------------------------------
// K3: attention row i: softmax over 1024 keys for 4 heads (hd=16);
//     writes attn_mean[i][:] and msg[i][:64]. wave w -> head (w&3), half (w>>2)
// ---------------------------------------------------------------------------
__global__ void k_attn(const float* __restrict__ qkv, float* __restrict__ attn_mean,
                       float* __restrict__ msg) {
  __shared__ float qs[64];
  __shared__ float am[1024];
  __shared__ float msgsh[64];
  __shared__ float hred[8];
  int t = threadIdx.x, lane = t & 31, w = t >> 5;
  int h = w & 3, half = w >> 2, jb = half * 512;
  int i = blockIdx.x;
  if (t < 64) { qs[t] = qkv[i * 192 + t]; msgsh[t] = 0.f; }
  for (int idx = t; idx < 1024; idx += 256) am[idx] = 0.f;
  __syncthreads();

  float sc[16];
#pragma unroll
  for (int kk = 0; kk < 16; ++kk) {
    int j = jb + kk * 32 + lane;
    const float* kp = qkv + j * 192 + 64 + h * 16;
    float a = 0.f;
#pragma unroll
    for (int d = 0; d < 16; ++d) a += qs[h * 16 + d] * kp[d];
    sc[kk] = a * 0.125f;   // 1/sqrt(64)
  }
  float lm = sc[0];
#pragma unroll
  for (int kk = 1; kk < 16; ++kk) lm = fmaxf(lm, sc[kk]);
#pragma unroll
  for (int m = 1; m < 32; m <<= 1) lm = fmaxf(lm, __shfl_xor(lm, m));
  if (lane == 0) hred[w] = lm;
  __syncthreads();
  float mx = fmaxf(hred[h], hred[h + 4]);
  __syncthreads();
  float ls = 0.f;
#pragma unroll
  for (int kk = 0; kk < 16; ++kk) { sc[kk] = __expf(sc[kk] - mx); ls += sc[kk]; }
#pragma unroll
  for (int m = 1; m < 32; m <<= 1) ls += __shfl_xor(ls, m);
  if (lane == 0) hred[w] = ls;
  __syncthreads();
  float inv = __builtin_amdgcn_rcpf(hred[h] + hred[h + 4]);

  float macc[16];
#pragma unroll
  for (int d = 0; d < 16; ++d) macc[d] = 0.f;
#pragma unroll
  for (int kk = 0; kk < 16; ++kk) {
    int j = jb + kk * 32 + lane;
    float a = sc[kk] * inv;
    atomicAdd(&am[j], a * 0.25f);
    const float* vp = qkv + j * 192 + 128 + h * 16;
#pragma unroll
    for (int d = 0; d < 16; ++d) macc[d] += a * vp[d];
  }
#pragma unroll
  for (int d = 0; d < 16; ++d) atomicAdd(&msgsh[h * 16 + d], macc[d]);
  __syncthreads();
  for (int idx = t; idx < 1024; idx += 256) attn_mean[i * 1024 + idx] = am[idx];
  if (t < 64) msg[i * 64 + t] = msgsh[t];
}

// ---------------------------------------------------------------------------
// K4: per-row tail: x += msg@out_w+b; x += GELU(LN2(x)@W1+b1)@W2+b2; gate(x)
// ---------------------------------------------------------------------------
__global__ void k_x(const float* __restrict__ x, const float* __restrict__ msg,
                    const float* __restrict__ out_w, const float* __restrict__ out_b,
                    const float* __restrict__ ln2_g, const float* __restrict__ ln2_b,
                    const float* __restrict__ ffn_w1, const float* __restrict__ ffn_b1,
                    const float* __restrict__ ffn_w2, const float* __restrict__ ffn_b2,
                    const float* __restrict__ gate_w1, const float* __restrict__ gate_b1,
                    const float* __restrict__ gate_w2, const float* __restrict__ gate_b2,
                    float* __restrict__ xout, float* __restrict__ gate) {
  __shared__ float xs[64], ms[64], x2s[64], hs[64], hid[256], red[64];
  int t = threadIdx.x, i = blockIdx.x;
  if (t < 64) { xs[t] = x[i * 64 + t]; ms[t] = msg[i * 64 + t]; }
  __syncthreads();
  if (t < 64) {
    float a = out_b[t];
#pragma unroll
    for (int m = 0; m < 64; ++m) a += ms[m] * out_w[m * 64 + t];
    x2s[t] = xs[t] + a;
    red[t] = x2s[t];
  }
  __syncthreads();
  for (int s = 32; s > 0; s >>= 1) { if (t < s) red[t] += red[t + s]; __syncthreads(); }
  float mean = red[0] * (1.f / 64.f);
  __syncthreads();
  if (t < 64) { float d = x2s[t] - mean; red[t] = d * d; }
  __syncthreads();
  for (int s = 32; s > 0; s >>= 1) { if (t < s) red[t] += red[t + s]; __syncthreads(); }
  float inv = rsqrtf(red[0] * (1.f / 64.f) + 1e-5f);
  __syncthreads();
  if (t < 64) hs[t] = (x2s[t] - mean) * inv * ln2_g[t] + ln2_b[t];
  __syncthreads();
  {
    float a = ffn_b1[t];
#pragma unroll
    for (int c = 0; c < 64; ++c) a += hs[c] * ffn_w1[c * 256 + t];
    hid[t] = 0.5f * a * (1.f + erff(a * 0.70710678f));   // exact GELU
  }
  __syncthreads();
  if (t < 64) {
    float o = ffn_b2[t];
    for (int k = 0; k < 256; ++k) o += hid[k] * ffn_w2[k * 64 + t];
    float xf = x2s[t] + o;
    xout[i * 64 + t] = xf;
    hs[t] = xf;
  }
  __syncthreads();
  if (t < 64) {
    float a = gate_b1[t];
#pragma unroll
    for (int m = 0; m < 64; ++m) a += hs[m] * gate_w1[m * 64 + t];
    red[t] = fast_silu(a) * gate_w2[t];
  }
  __syncthreads();
  for (int s = 32; s > 0; s >>= 1) { if (t < s) red[t] += red[t + s]; __syncthreads(); }
  if (t == 0) gate[i] = fast_sigmoid(red[0] + gate_b2[0]);
}

// ---------------------------------------------------------------------------
// K5: pairwise WMMA stage. Wave owns row i; j chunk = blockIdx.y*256.
//     Per 16-pair tile: enc -> WMMA(16x16x32 f16, K pad 9->32) -> silu -> LDS
//     -> WMMA (K=64, Wc composed) -> silu -> dot(coord_w2) -> *attn_mean
//     -> delta += cw * rel
// ---------------------------------------------------------------------------
__global__ void __launch_bounds__(256, 1)
k_pair(const float* __restrict__ coors, const float* __restrict__ attn_mean,
       const _Float16* __restrict__ WcT, const _Float16* __restrict__ w1T,
       const float* __restrict__ bc, const float* __restrict__ edge_b1,
       const float* __restrict__ coord_w2, const float* __restrict__ coord_b2,
       float* __restrict__ delta4) {
  __shared__ __align__(32) _Float16 h2buf[8][16][64];  // per-wave 16x64 fp16 (silu(h1))
  __shared__ float cwbuf[8][16];

  const int t = threadIdx.x, lane = t & 31, wv = t >> 5;
  const int p = lane & 15, hi = lane >> 4;
  const int i = blockIdx.x * 8 + wv;
  const int jbase = blockIdx.y * 256;

  // B fragments (column n = lane&15 (+16t); element e -> K = hi*16 + e [+kb*32])
  v16h zf;
#pragma unroll
  for (int e = 0; e < 16; ++e) zf[e] = (_Float16)0.f;
  v16h b1f[4], b2f[4][2];
#pragma unroll
  for (int tt = 0; tt < 4; ++tt) {
    int col = tt * 16 + p;
    b1f[tt] = hi ? zf : *(const v16h*)(w1T + col * 16);           // K>=16 all zero
    b2f[tt][0] = *(const v16h*)(WcT + col * 64 + hi * 16);
    b2f[tt][1] = *(const v16h*)(WcT + col * 64 + 32 + hi * 16);
  }
  float b1c[4], bcc[4], w2c[4];
#pragma unroll
  for (int tt = 0; tt < 4; ++tt) {
    b1c[tt] = edge_b1[tt * 16 + p];
    bcc[tt] = bc[tt * 16 + p];
    w2c[tt] = coord_w2[tt * 16 + p];
  }
  const float cb2 = coord_b2[0];
  const float cix = coors[i * 3 + 0], ciy = coors[i * 3 + 1], ciz = coors[i * 3 + 2];
  float dax = 0.f, day = 0.f, daz = 0.f;

  for (int jt = 0; jt < 16; ++jt) {
    const int j = jbase + jt * 16 + p;
    const float rx = cix - coors[j * 3 + 0];
    const float ry = ciy - coors[j * 3 + 1];
    const float rz = ciz - coors[j * 3 + 2];
    const float rd = rx * rx + ry * ry + rz * rz;
    const float s0 = __sinf(rd), s1 = __sinf(rd * 0.5f), s2 = __sinf(rd * 0.25f), s3 = __sinf(rd * 0.125f);
    const float c0 = __cosf(rd), c1 = __cosf(rd * 0.5f), c2 = __cosf(rd * 0.25f), c3 = __cosf(rd * 0.125f);

    // A1 fragment (row = p). lanes hi=0 carry K=0..8 (sin,cos,rd at element 8); hi=1 zero
    v16h a1 = zf;
    a1[0] = (_Float16)(hi ? 0.f : s0);
    a1[1] = (_Float16)(hi ? 0.f : s1);
    a1[2] = (_Float16)(hi ? 0.f : s2);
    a1[3] = (_Float16)(hi ? 0.f : s3);
    a1[4] = (_Float16)(hi ? 0.f : c0);
    a1[5] = (_Float16)(hi ? 0.f : c1);
    a1[6] = (_Float16)(hi ? 0.f : c2);
    a1[7] = (_Float16)(hi ? 0.f : c3);
    a1[8] = (_Float16)(hi ? 0.f : rd);

    // GEMM1: enc @ w1 + b1, then silu -> LDS
#pragma unroll
    for (int tt = 0; tt < 4; ++tt) {
      v8f acc;
#pragma unroll
      for (int e = 0; e < 8; ++e) acc[e] = b1c[tt];
      acc = __builtin_amdgcn_wmma_f32_16x16x32_f16(false, a1, false, b1f[tt], (short)0, acc, false, false);
      int col = tt * 16 + p;
#pragma unroll
      for (int r = 0; r < 8; ++r) {
        h2buf[wv][r + hi * 8][col] = (_Float16)fast_silu(acc[r]);
      }
    }
    __syncthreads();

    // A2 fragments from LDS (row = p; element e -> K = hi*16 + e, matching B)
    v8h q0 = *(const v8h*)&h2buf[wv][p][hi * 16];
    v8h q1 = *(const v8h*)&h2buf[wv][p][hi * 16 + 8];
    v8h q2 = *(const v8h*)&h2buf[wv][p][32 + hi * 16];
    v8h q3 = *(const v8h*)&h2buf[wv][p][40 + hi * 16];
    v16h a2k0 = __builtin_shufflevector(q0, q1, 0, 1, 2, 3, 4, 5, 6, 7, 8, 9, 10, 11, 12, 13, 14, 15);
    v16h a2k1 = __builtin_shufflevector(q2, q3, 0, 1, 2, 3, 4, 5, 6, 7, 8, 9, 10, 11, 12, 13, 14, 15);

    // GEMM2: h1 @ Wc + bc, silu, dot coord_w2
    float part[8];
#pragma unroll
    for (int r = 0; r < 8; ++r) part[r] = 0.f;
#pragma unroll
    for (int tt = 0; tt < 4; ++tt) {
      v8f acc;
#pragma unroll
      for (int e = 0; e < 8; ++e) acc[e] = bcc[tt];
      acc = __builtin_amdgcn_wmma_f32_16x16x32_f16(false, a2k0, false, b2f[tt][0], (short)0, acc, false, false);
      acc = __builtin_amdgcn_wmma_f32_16x16x32_f16(false, a2k1, false, b2f[tt][1], (short)0, acc, false, false);
#pragma unroll
      for (int r = 0; r < 8; ++r) {
        part[r] += fast_silu(acc[r]) * w2c[tt];
      }
    }
    // reduce across the 16 lanes of each half-group (rows hi*8 + r)
#pragma unroll
    for (int r = 0; r < 8; ++r) {
#pragma unroll
      for (int m = 1; m < 16; m <<= 1) part[r] += __shfl_xor(part[r], m);
    }
    if (p < 8) {
      float val = part[0];
#pragma unroll
      for (int r = 1; r < 8; ++r) val = (p == r) ? part[r] : val;
      cwbuf[wv][hi * 8 + p] = val + cb2;
    }
    __syncthreads();

    float cw = cwbuf[wv][p] * attn_mean[i * 1024 + j];
    if (hi == 0) { dax += cw * rx; day += cw * ry; daz += cw * rz; }
  }
#pragma unroll
  for (int m = 1; m < 32; m <<= 1) {
    dax += __shfl_xor(dax, m);
    day += __shfl_xor(day, m);
    daz += __shfl_xor(daz, m);
  }
  if (lane == 0) {
    float* dst = delta4 + blockIdx.y * (N_TOK * 3) + i * 3;
    dst[0] = dax; dst[1] = day; dst[2] = daz;
  }
}

// ---------------------------------------------------------------------------
// K6: coors_out = coors + (sum of chunk deltas) * gate
// ---------------------------------------------------------------------------
__global__ void k_coor(const float* __restrict__ coors, const float* __restrict__ delta4,
                       const float* __restrict__ gate, float* __restrict__ out) {
  int idx = blockIdx.x * 256 + threadIdx.x;  // 0..3071
  int i = idx / 3;
  float d = delta4[idx] + delta4[3072 + idx] + delta4[6144 + idx] + delta4[9216 + idx];
  out[idx] = coors[idx] + d * gate[i];
}

// ---------------------------------------------------------------------------
extern "C" void kernel_launch(void* const* d_in, const int* in_sizes, int n_in,
                              void* d_out, int out_size, void* d_ws, size_t ws_size,
                              hipStream_t stream) {
  const float* feats    = (const float*)d_in[0];
  const float* coors    = (const float*)d_in[1];
  const float* qkv_w    = (const float*)d_in[2];
  const float* out_w    = (const float*)d_in[3];
  const float* out_b    = (const float*)d_in[4];
  const float* edge_w1  = (const float*)d_in[5];
  const float* edge_b1  = (const float*)d_in[6];
  const float* edge_w2  = (const float*)d_in[7];
  const float* edge_b2  = (const float*)d_in[8];
  const float* coord_w1 = (const float*)d_in[9];
  const float* coord_b1 = (const float*)d_in[10];
  const float* coord_w2 = (const float*)d_in[11];
  const float* coord_b2 = (const float*)d_in[12];
  const float* gate_w1  = (const float*)d_in[13];
  const float* gate_b1  = (const float*)d_in[14];
  const float* gate_w2  = (const float*)d_in[15];
  const float* gate_b2  = (const float*)d_in[16];
  const float* ln1_g    = (const float*)d_in[17];
  const float* ln1_b    = (const float*)d_in[18];
  const float* ln2_g    = (const float*)d_in[19];
  const float* ln2_b    = (const float*)d_in[20];
  const float* ffn_w1   = (const float*)d_in[21];
  const float* ffn_b1   = (const float*)d_in[22];
  const float* ffn_w2   = (const float*)d_in[23];
  const float* ffn_b2   = (const float*)d_in[24];

  float* attn_mean = (float*)d_ws;                 // 1024*1024
  float* x      = attn_mean + 1024 * 1024;         // 1024*64
  float* qkv    = x + 1024 * 64;                   // 1024*192
  float* msg    = qkv + 1024 * 192;                // 1024*64
  float* gate   = msg + 1024 * 64;                 // 1024
  float* delta4 = gate + 1024;                     // 4*1024*3
  float* bc     = delta4 + 4 * 1024 * 3;           // 64
  _Float16* WcT = (_Float16*)(bc + 64);            // 64*64 fp16 (32B aligned)
  _Float16* w1T = WcT + 64 * 64;                   // 64*16 fp16

  float* xout    = (float*)d_out;                  // 1024*64
  float* coorout = xout + 1024 * 64;               // 1024*3

  k_prepw<<<1, 256, 0, stream>>>(edge_w2, coord_w1, edge_b2, coord_b1, edge_w1, WcT, bc, w1T);
  k_ln1  <<<128, 256, 0, stream>>>(feats, ln1_g, ln1_b, x);
  k_qkv  <<<1024, 256, 0, stream>>>(x, qkv_w, qkv);
  k_attn <<<1024, 256, 0, stream>>>(qkv, attn_mean, msg);
  k_x    <<<1024, 256, 0, stream>>>(x, msg, out_w, out_b, ln2_g, ln2_b,
                                    ffn_w1, ffn_b1, ffn_w2, ffn_b2,
                                    gate_w1, gate_b1, gate_w2, gate_b2, xout, gate);
  k_pair <<<dim3(128, 4), 256, 0, stream>>>(coors, attn_mean, WcT, w1T, bc, edge_b1,
                                            coord_w2, coord_b2, delta4);
  k_coor <<<12, 256, 0, stream>>>(coors, delta4, gate, coorout);
}
